// MoELayer_57724360458723
// MI455X (gfx1250) — matrix operand order repaired and checked by reading, not compiled, not run on hardware
//
#include <hip/hip_runtime.h>

// MoE dense-expert block for MI455X (gfx1250, wave32, WMMA + async-LDS).
// Compute-bound (~550 GFLOP f32 GEMM vs ~200MB HBM traffic @23.3TB/s).
// Strategy: split-bf16 (hi + lo residual) GEMMs on v_wmma_f32_16x16x32_bf16:
//   A*B ~= Ah*Bh + Ah*Bl + Al*Bh   (error ~2^-16 relative, fp32-grade)
// Tiles staged through LDS with double-buffered GLOBAL_LOAD_ASYNC_TO_LDS_B128
// (ASYNCcnt-tracked), removing intra-workgroup operand re-fetch from L2.

#define HDIM 1024
#define NTOK 16384
#define NEXP 8

typedef __attribute__((ext_vector_type(8)))  __bf16 v8bf;
typedef __attribute__((ext_vector_type(16))) __bf16 v16bf;
typedef __attribute__((ext_vector_type(8)))  float  v8f;

// ---------------- CDNA5 async copy: global -> LDS, 16B per lane ----------------
__device__ __forceinline__ void async_b128(unsigned lds_byte, const void* base,
                                           unsigned voff_byte) {
  asm volatile("global_load_async_to_lds_b128 %0, %1, %2"
               :
               : "v"(lds_byte), "v"(voff_byte), "s"((unsigned long long)base)
               : "memory");
}

// ---------------- preprocessing: f32 -> bf16 hi/lo split ----------------
__global__ __launch_bounds__(256) void moe_split(const float* __restrict__ src,
                                                 __bf16* __restrict__ hi,
                                                 __bf16* __restrict__ lo, int n) {
  int i = blockIdx.x * 256 + threadIdx.x;
  if (i < n) {
    float v = src[i];
    __bf16 h = (__bf16)v;
    hi[i] = h;
    lo[i] = (__bf16)(v - (float)h);
  }
}

// Transpose W[e][i][o] -> Wt[e][o][i], split into bf16 hi/lo. LDS-tiled 32x32.
__global__ __launch_bounds__(256) void moe_wsplit_tr(const float* __restrict__ W,
                                                     __bf16* __restrict__ hi,
                                                     __bf16* __restrict__ lo) {
  __shared__ float tile[32][33];
  int e = blockIdx.z;
  int i0 = blockIdx.x * 32;
  int o0 = blockIdx.y * 32;
  int tx = threadIdx.x, ty = threadIdx.y;   // (32, 8)
  const float* Wb = W + (size_t)e * HDIM * HDIM;
#pragma unroll
  for (int r = ty; r < 32; r += 8)
    tile[r][tx] = Wb[(size_t)(i0 + r) * HDIM + (o0 + tx)];
  __syncthreads();
  __bf16* hib = hi + (size_t)e * HDIM * HDIM;
  __bf16* lob = lo + (size_t)e * HDIM * HDIM;
#pragma unroll
  for (int r = ty; r < 32; r += 8) {
    float v = tile[tx][r];                               // = W[i0+tx][o0+r]
    __bf16 h = (__bf16)v;
    size_t o = (size_t)(o0 + r) * HDIM + (i0 + tx);      // Wt[o][i]
    hib[o] = h;
    lob[o] = (__bf16)(v - (float)h);
  }
}

// ---------------- gating: logits, softmax, top-2, renormalize ----------------
__global__ __launch_bounds__(256) void moe_gate(const float* __restrict__ x,
                                                const float* __restrict__ Wg,
                                                const float* __restrict__ bg,
                                                float* __restrict__ gate_out,
                                                float* __restrict__ wcomb) {
  int lane = threadIdx.x & 31;
  int wave = threadIdx.x >> 5;
  int n = blockIdx.x * 8 + wave;            // one wave32 per token
  float p[NEXP];
#pragma unroll
  for (int e = 0; e < NEXP; ++e) p[e] = 0.f;
  const float* xr = x + (size_t)n * HDIM;
  for (int h = lane; h < HDIM; h += 32) {
    float xv = xr[h];
    const float* wr = Wg + (size_t)h * NEXP;
#pragma unroll
    for (int e = 0; e < NEXP; ++e) p[e] += xv * wr[e];
  }
#pragma unroll
  for (int e = 0; e < NEXP; ++e)
    for (int off = 16; off > 0; off >>= 1) p[e] += __shfl_down(p[e], off, 32);
  if (lane == 0) {
    float mx = -1e30f;
#pragma unroll
    for (int e = 0; e < NEXP; ++e) { p[e] += bg[e]; mx = fmaxf(mx, p[e]); }
    float sum = 0.f;
#pragma unroll
    for (int e = 0; e < NEXP; ++e) { p[e] = expf(p[e] - mx); sum += p[e]; }
    float inv = 1.f / sum;
#pragma unroll
    for (int e = 0; e < NEXP; ++e) { p[e] *= inv; gate_out[(size_t)n * NEXP + e] = p[e]; }
    int i0 = 0;
#pragma unroll
    for (int e = 1; e < NEXP; ++e) if (p[e] > p[i0]) i0 = e;
    int i1 = (i0 == 0) ? 1 : 0;
#pragma unroll
    for (int e = 0; e < NEXP; ++e) if (e != i0 && p[e] > p[i1]) i1 = e;
    float denom = p[i0] + p[i1] + 1e-9f;
    float w0 = p[i0] / denom, w1 = p[i1] / denom;
#pragma unroll
    for (int e = 0; e < NEXP; ++e)
      wcomb[(size_t)n * NEXP + e] = (e == i0) ? w0 : (e == i1) ? w1 : 0.f;
  }
}

// ---------------- split-bf16 WMMA GEMM with async-LDS double buffering ----------------
// C[M,1024] = A[M,1024] @ Bt^T, Bt = [1024(out),1024(in)] pre-transposed weights.
// Block tile 64(M) x 128(N), K-step 32. 8 waves in 2x4; wave owns 2x2 16x16 tiles.
// LDS rows padded to 80B (40 bf16) for aligned, low-conflict ds_load_b128.
#define LSTR   40                       // bf16 elems per LDS row (32 data + 8 pad)
#define AH_OFF 0                        // byte offsets inside one LDS buffer
#define AL_OFF 5120                     // 64*80
#define BH_OFF 10240
#define BL_OFF 20480                    // BH + 128*80
#define BUF_BYTES 30720                 // 2*5120 + 2*10240
#define BUF_ELEMS 15360

__global__ __launch_bounds__(256) void moe_gemm_bf16x2(
    const __bf16* __restrict__ Ah, const __bf16* __restrict__ Al,
    const __bf16* __restrict__ Bh, const __bf16* __restrict__ Bl,
    float* __restrict__ C) {
  __shared__ __align__(16) __bf16 smem[2 * BUF_ELEMS];   // 60KB

  const int t    = threadIdx.x;
  const int lane = t & 31;
  const int wave = t >> 5;
  const int wm = wave >> 2, wn = wave & 3;
  const int l16 = lane & 15, lh = lane >> 4;
  const int rowBase0 = blockIdx.y * 64;
  const int colBase0 = blockIdx.x * 128;

  // --- async-copy thread mapping: 256 threads; t -> (row, 16B k-chunk) ---
  const int arow = t >> 2;              // 0..63
  const int kc   = t & 3;               // 0..3 (16B chunks of the K=32 slab)
  const unsigned smem_base = (unsigned)(size_t)(&smem[0]);
  const unsigned aLds = (unsigned)(arow * 80 + kc * 16);
  const unsigned aGlob = (unsigned)((rowBase0 + arow) * (HDIM * 2) + kc * 16);
  unsigned bLds[2], bGlob[2];
#pragma unroll
  for (int j = 0; j < 2; ++j) {
    int br = 64 * j + arow;             // 0..127 rows of the B tile
    bLds[j]  = (unsigned)(br * 80 + kc * 16);
    bGlob[j] = (unsigned)((colBase0 + br) * (HDIM * 2) + kc * 16);
  }

  auto issue = [&](int buf, int k) {
    const unsigned kb = (unsigned)k * 2;
    const unsigned lb = smem_base + (unsigned)buf * BUF_BYTES;
    async_b128(lb + AH_OFF + aLds, Ah, aGlob + kb);
    async_b128(lb + AL_OFF + aLds, Al, aGlob + kb);
#pragma unroll
    for (int j = 0; j < 2; ++j) {
      async_b128(lb + BH_OFF + bLds[j], Bh, bGlob[j] + kb);
      async_b128(lb + BL_OFF + bLds[j], Bl, bGlob[j] + kb);
    }
  };

  v8f acc[2][2] = {};

  issue(0, 0);                          // prologue: first K-slab
  int cur = 0;
  for (int i = 0; i < HDIM / 32; ++i) {
    if (i + 1 < HDIM / 32) {
      issue(cur ^ 1, (i + 1) * 32);     // prefetch next slab into alternate buffer
      asm volatile("s_wait_asynccnt 6" ::: "memory");  // current slab done (in-order)
    } else {
      asm volatile("s_wait_asynccnt 0" ::: "memory");
    }
    __syncthreads();                    // all waves' transfers visible

    const __bf16* sb = &smem[cur * BUF_ELEMS];
    v16bf ahf[2], alf[2];
#pragma unroll
    for (int mf = 0; mf < 2; ++mf) {
      int row = wm * 32 + mf * 16 + l16;
      const __bf16* pa = sb + (AH_OFF / 2) + row * LSTR + lh * 8;
      v8bf a0 = *(const v8bf*)pa;
      v8bf a1 = *(const v8bf*)(pa + 16);
      ahf[mf] = __builtin_shufflevector(a0, a1, 0,1,2,3,4,5,6,7,8,9,10,11,12,13,14,15);
      const __bf16* pl = sb + (AL_OFF / 2) + row * LSTR + lh * 8;
      v8bf c0 = *(const v8bf*)pl;
      v8bf c1 = *(const v8bf*)(pl + 16);
      alf[mf] = __builtin_shufflevector(c0, c1, 0,1,2,3,4,5,6,7,8,9,10,11,12,13,14,15);
    }
#pragma unroll
    for (int nf = 0; nf < 2; ++nf) {
      int brow = wn * 32 + nf * 16 + l16;
      const __bf16* pb = sb + (BH_OFF / 2) + brow * LSTR + lh * 8;
      v8bf b0 = *(const v8bf*)pb;
      v8bf b1 = *(const v8bf*)(pb + 16);
      v16bf bhf = __builtin_shufflevector(b0, b1, 0,1,2,3,4,5,6,7,8,9,10,11,12,13,14,15);
      const __bf16* ql = sb + (BL_OFF / 2) + brow * LSTR + lh * 8;
      v8bf d0 = *(const v8bf*)ql;
      v8bf d1 = *(const v8bf*)(ql + 16);
      v16bf blf = __builtin_shufflevector(d0, d1, 0,1,2,3,4,5,6,7,8,9,10,11,12,13,14,15);
#pragma unroll
      for (int mf = 0; mf < 2; ++mf) {
        acc[mf][nf] = __builtin_amdgcn_wmma_f32_16x16x32_bf16(
            false, ahf[mf], false, bhf, (short)0, acc[mf][nf], false, false);
        acc[mf][nf] = __builtin_amdgcn_wmma_f32_16x16x32_bf16(
            false, ahf[mf], false, blf, (short)0, acc[mf][nf], false, false);
        acc[mf][nf] = __builtin_amdgcn_wmma_f32_16x16x32_bf16(
            false, alf[mf], false, bhf, (short)0, acc[mf][nf], false, false);
      }
    }
    __syncthreads();                    // done reading buf before it is overwritten
    cur ^= 1;
  }

  // C/D layout: VGPR i -> row = i + 8*(lane>=16), col = lane%16
#pragma unroll
  for (int mf = 0; mf < 2; ++mf)
#pragma unroll
    for (int nf = 0; nf < 2; ++nf) {
      int r0 = rowBase0 + wm * 32 + mf * 16 + lh * 8;
      int c  = colBase0 + wn * 32 + nf * 16 + l16;
#pragma unroll
      for (int i = 0; i < 8; ++i)
        C[(size_t)(r0 + i) * HDIM + c] = acc[mf][nf][i];
    }
}

// ---------------- row statistics helper (256 threads, one row of 1024) ----------------
__device__ __forceinline__ void row_stats(float s, float s2, float& mean, float& rstd) {
#pragma unroll
  for (int off = 16; off > 0; off >>= 1) {
    s  += __shfl_down(s, off, 32);
    s2 += __shfl_down(s2, off, 32);
  }
  __shared__ float rs[8], rs2[8], stat[2];
  int tid = threadIdx.x, w = tid >> 5, lane = tid & 31;
  if (lane == 0) { rs[w] = s; rs2[w] = s2; }
  __syncthreads();
  if (tid == 0) {
    float S = 0.f, S2 = 0.f;
#pragma unroll
    for (int i = 0; i < 8; ++i) { S += rs[i]; S2 += rs2[i]; }
    float m = S / (float)HDIM;
    float var = S2 / (float)HDIM - m * m;
    stat[0] = m;
    stat[1] = rsqrtf(var + 1e-5f);
  }
  __syncthreads();
  mean = stat[0];
  rstd = stat[1];
}

// h1 = relu(LN(gemm1 + b1)); re-split to bf16 hi/lo for the second GEMM.
__global__ __launch_bounds__(256) void moe_ln_relu_split(
    const float* __restrict__ tmp, const float* __restrict__ b1,
    const float* __restrict__ g1, const float* __restrict__ be1, int e,
    __bf16* __restrict__ Hh, __bf16* __restrict__ Hl) {
  int n = blockIdx.x, tid = threadIdx.x;
  const float* row = tmp + (size_t)n * HDIM;
  const float* bb = b1 + (size_t)e * HDIM;
  const float* gg = g1 + (size_t)e * HDIM;
  const float* bt = be1 + (size_t)e * HDIM;
  float vals[4], s = 0.f, s2 = 0.f;
#pragma unroll
  for (int j = 0; j < 4; ++j) {
    int h = tid + j * 256;
    float v = row[h] + bb[h];
    vals[j] = v; s += v; s2 += v * v;
  }
  float m, r;
  row_stats(s, s2, m, r);
#pragma unroll
  for (int j = 0; j < 4; ++j) {
    int h = tid + j * 256;
    float v = (vals[j] - m) * r * gg[h] + bt[h];
    v = fmaxf(v, 0.f);
    __bf16 hv = (__bf16)v;
    size_t o = (size_t)n * HDIM + h;
    Hh[o] = hv;
    Hl[o] = (__bf16)(v - (float)hv);
  }
}

// y = relu(x + LN(gemm2 + b2)); out (+)= wcomb[n,e] * y   (init at e==0)
__global__ __launch_bounds__(256) void moe_combine(
    const float* __restrict__ tmp, const float* __restrict__ x,
    const float* __restrict__ b2, const float* __restrict__ g2,
    const float* __restrict__ be2, const float* __restrict__ wcomb,
    float* __restrict__ out, int e) {
  int n = blockIdx.x, tid = threadIdx.x;
  const float* row = tmp + (size_t)n * HDIM;
  const float* xr = x + (size_t)n * HDIM;
  const float* bb = b2 + (size_t)e * HDIM;
  const float* gg = g2 + (size_t)e * HDIM;
  const float* bt = be2 + (size_t)e * HDIM;
  float vals[4], s = 0.f, s2 = 0.f;
#pragma unroll
  for (int j = 0; j < 4; ++j) {
    int h = tid + j * 256;
    float v = row[h] + bb[h];
    vals[j] = v; s += v; s2 += v * v;
  }
  float m, r;
  row_stats(s, s2, m, r);
  float wgt = wcomb[(size_t)n * NEXP + e];
#pragma unroll
  for (int j = 0; j < 4; ++j) {
    int h = tid + j * 256;
    float v = (vals[j] - m) * r * gg[h] + bt[h];
    float y = fmaxf(xr[h] + v, 0.f);
    size_t o = (size_t)n * HDIM + h;
    out[o] = (e == 0) ? wgt * y : out[o] + wgt * y;
  }
}

// ---------------- launcher ----------------
extern "C" void kernel_launch(void* const* d_in, const int* in_sizes, int n_in,
                              void* d_out, int out_size, void* d_ws, size_t ws_size,
                              hipStream_t stream) {
  (void)in_sizes; (void)n_in; (void)out_size; (void)ws_size;
  const float* x   = (const float*)d_in[0];
  const float* Wg  = (const float*)d_in[1];
  const float* bg  = (const float*)d_in[2];
  const float* W1  = (const float*)d_in[3];
  const float* b1  = (const float*)d_in[4];
  const float* g1  = (const float*)d_in[5];
  const float* be1 = (const float*)d_in[6];
  const float* W2  = (const float*)d_in[7];
  const float* b2  = (const float*)d_in[8];
  const float* g2  = (const float*)d_in[9];
  const float* be2 = (const float*)d_in[10];

  float* out = (float*)d_out;                       // moe_out [N,H]
  float* gate_out = out + (size_t)NTOK * HDIM;      // gate_probs [N,E]

  const size_t NH  = (size_t)NTOK * HDIM;
  const size_t HH  = (size_t)HDIM * HDIM;
  const size_t EHH = (size_t)NEXP * HH;

  char* ws = (char*)d_ws;
  __bf16* Xh   = (__bf16*)ws; ws += NH * 2;
  __bf16* Xl   = (__bf16*)ws; ws += NH * 2;
  __bf16* Hh   = (__bf16*)ws; ws += NH * 2;
  __bf16* Hl   = (__bf16*)ws; ws += NH * 2;
  __bf16* W1th = (__bf16*)ws; ws += EHH * 2;
  __bf16* W1tl = (__bf16*)ws; ws += EHH * 2;
  __bf16* W2th = (__bf16*)ws; ws += EHH * 2;
  __bf16* W2tl = (__bf16*)ws; ws += EHH * 2;
  float*  tmp   = (float*)ws; ws += NH * 4;
  float*  wcomb = (float*)ws;

  // preprocessing
  moe_split<<<(unsigned)(NH / 256), 256, 0, stream>>>(x, Xh, Xl, (int)NH);
  dim3 tb(32, 8), tg(HDIM / 32, HDIM / 32, NEXP);
  moe_wsplit_tr<<<tg, tb, 0, stream>>>(W1, W1th, W1tl);
  moe_wsplit_tr<<<tg, tb, 0, stream>>>(W2, W2th, W2tl);
  moe_gate<<<NTOK / 8, 256, 0, stream>>>(x, Wg, bg, gate_out, wcomb);

  dim3 gg(HDIM / 128, NTOK / 64);   // (8, 256) workgroups of 256 threads (8 waves)
  for (int e = 0; e < NEXP; ++e) {
    const size_t wo = (size_t)e * HH;
    moe_gemm_bf16x2<<<gg, 256, 0, stream>>>(Xh, Xl, W1th + wo, W1tl + wo, tmp);
    moe_ln_relu_split<<<NTOK, 256, 0, stream>>>(tmp, b1, g1, be1, e, Hh, Hl);
    moe_gemm_bf16x2<<<gg, 256, 0, stream>>>(Hh, Hl, W2th + wo, W2tl + wo, tmp);
    moe_combine<<<NTOK, 256, 0, stream>>>(tmp, x, b2, g2, be2, wcomb, out, e);
  }
}
